// LlamaAttention_49383533969947
// MI455X (gfx1250) — matrix operand order repaired and checked by reading, not compiled, not run on hardware
//
#include <hip/hip_runtime.h>
#include <hip/hip_bf16.h>
#include <math.h>

// ---------------- problem constants ----------------
constexpr int Ts  = 2048;          // sequence length
constexpr int Dd  = 2048;          // model dim
constexpr int NH  = 32;            // query heads
constexpr int KV  = 8;             // kv heads
constexpr int HD  = 64;            // head dim
constexpr int Gg  = NH / KV;       // 4
constexpr int KVD = KV * HD;       // 512
constexpr int NHD = NH * HD;       // 2048
constexpr float THETA   = 500000.0f;
constexpr float FACTOR  = 32.0f;
constexpr float LOW_F   = 1.0f;
constexpr float HIGH_F  = 4.0f;
constexpr float OLD_CTX = 8192.0f;

typedef __bf16 bf16_t;
typedef __attribute__((ext_vector_type(16))) __bf16 v16bf;
typedef __attribute__((ext_vector_type(8)))  __bf16 v8bf;
typedef __attribute__((ext_vector_type(8)))  float  v8f;

// ---------------- async-to-LDS staging ----------------
// The clang builtin for the async load has a v4i descriptor-style signature
// (probe: "cannot initialize a parameter of type 'int __vector(4)'"), so we
// emit the ISA instruction directly. VGLOBAL encoding (ISA 15.18):
//   global_load_async_to_lds_b128 vdst, vaddr, off
// where vdst = per-lane LDS byte address,
// vaddr = 64-bit global address. Tracked by ASYNCcnt.
#define USE_ASYNC_LDS 1

__device__ __forceinline__ void stage16(const bf16_t* g, bf16_t* l) {
#if USE_ASYNC_LDS
    // Flat aperture maps LDS as addr[31:0] (ISA 10.2) -> low 32 bits of the
    // generic pointer to a __shared__ object are the LDS byte offset.
    unsigned laddr = (unsigned)(size_t)l;
    asm volatile("global_load_async_to_lds_b128 %0, %1, off"
                 :
                 : "v"(laddr), "v"(g)
                 : "memory");
#else
    *(v8bf*)l = *(const v8bf*)g;
#endif
}

__device__ __forceinline__ void wait_stage() {
#if USE_ASYNC_LDS
#if defined(__has_builtin) && __has_builtin(__builtin_amdgcn_s_wait_asynccnt)
    __builtin_amdgcn_s_wait_asynccnt(0);
#else
    asm volatile("s_wait_asynccnt 0x0" ::: "memory");
#endif
#endif
    __syncthreads();
}

// Within-wave LDS store->load ordering (DS is in-order per wave; block compiler reorder)
__device__ __forceinline__ void lds_fence_wave() {
    asm volatile("" ::: "memory");
#if defined(__has_builtin)
#if __has_builtin(__builtin_amdgcn_s_wait_dscnt)
    __builtin_amdgcn_s_wait_dscnt(0);
#endif
#endif
}

// ---------------------------------------------------------------------------
// Fragment loaders (wave32 WMMA layouts, CDNA5 ISA 7.12.2)
// A (16x32 bf16): lane l holds row m=l&15; K(e) = 8*(l>>4) + (e<8 ? e : e+8)
//                 -> two contiguous 8-element (16B) groups.
// B (32x16 bf16): lane l holds col n=l&15; K(e) = 8*(l>>4) + (e<8 ? e : e+8)
//                 -> two contiguous 8-element (16B) groups.
// C/D (16x16 f32): VGPR r, lane l -> row m = r + 8*(l>>4), col n = l&15.
// ---------------------------------------------------------------------------
__device__ __forceinline__ v16bf load_frag_a(const bf16_t* rowbase, int k0, int hi) {
    v8bf lo = *(const v8bf*)(rowbase + k0 + 8 * hi);
    v8bf hv = *(const v8bf*)(rowbase + k0 + 16 + 8 * hi);
    return __builtin_shufflevector(lo, hv, 0,1,2,3,4,5,6,7,8,9,10,11,12,13,14,15);
}

__device__ __forceinline__ v16bf load_frag_b2(const bf16_t* base, int hi) {
    v8bf lo = *(const v8bf*)(base + 8 * hi);
    v8bf hv = *(const v8bf*)(base + 16 + 8 * hi);
    return __builtin_shufflevector(lo, hv, 0,1,2,3,4,5,6,7,8,9,10,11,12,13,14,15);
}

__device__ __forceinline__ v8f wmma_bf16(v16bf a, v16bf b, v8f c) {
    return __builtin_amdgcn_wmma_f32_16x16x32_bf16(false, a, false, b, (short)0, c,
                                                   false, false);
}

// ---------------------------------------------------------------------------
// Elementwise convert f32 -> bf16
// ---------------------------------------------------------------------------
__global__ void cvt_f32_bf16(const float* __restrict__ src, bf16_t* __restrict__ dst,
                             int n) {
    int i = blockIdx.x * blockDim.x + threadIdx.x;
    if (i < n) dst[i] = (bf16_t)src[i];
}

// ---------------------------------------------------------------------------
// Transpose + convert:  src f32 [K][N]  ->  dst bf16 [N][K]
// ---------------------------------------------------------------------------
__global__ void tr_f32_bf16(const float* __restrict__ src, bf16_t* __restrict__ dst,
                            int K, int N) {
    int i = blockIdx.x * blockDim.x + threadIdx.x;
    if (i >= K * N) return;
    int k = i / N;
    int n = i - k * N;
    dst[(size_t)n * K + k] = (bf16_t)src[i];
}

// ---------------------------------------------------------------------------
// LDS-tiled bf16 GEMM:  C[M][N] f32 = A[M][K] bf16 x BT[N][K] bf16
// Block tile 128x128, BK=32, double-buffered async staging.
// 8 waves, each computing 16(M) x 128(N): 8 WMMAs per k-step, A-frag reused 8x.
// ---------------------------------------------------------------------------
constexpr int GP = 40;  // LDS row pitch (elements) for 32-element rows

__global__ void __launch_bounds__(256)
gemm_bf16_lds(const bf16_t* __restrict__ A, const bf16_t* __restrict__ BT,
              float* __restrict__ C, int M, int N, int K) {
    __shared__ bf16_t Al[2][128 * GP];
    __shared__ bf16_t Bl[2][128 * GP];

    const int tid  = threadIdx.x;
    const int w    = tid >> 5;
    const int lane = tid & 31;
    const int lo16 = lane & 15;
    const int hi   = lane >> 4;
    const int nBN  = N >> 7;
    const int bm   = blockIdx.x / nBN;
    const int bn   = blockIdx.x - bm * nBN;
    const int m0   = bm << 7;
    const int n0   = bn << 7;

    const int cRow = tid >> 2;            // staging row (0..63 per pass, x2 passes)
    const int cOff = (tid & 3) * 8;       // element offset within 32-wide row

    auto stage = [&](int kb, int b) {
        int k0 = kb << 5;
#pragma unroll
        for (int it = 0; it < 2; ++it) {
            int row = cRow + 64 * it;
            stage16(A  + (size_t)(m0 + row) * K + k0 + cOff, &Al[b][row * GP + cOff]);
            stage16(BT + (size_t)(n0 + row) * K + k0 + cOff, &Bl[b][row * GP + cOff]);
        }
    };

    v8f acc[8] = {v8f{}, v8f{}, v8f{}, v8f{}, v8f{}, v8f{}, v8f{}, v8f{}};
    const int nK = K >> 5;

    stage(0, 0);
    wait_stage();
    for (int kb = 0; kb < nK; ++kb) {
        int b = kb & 1;
        if (kb + 1 < nK) stage(kb + 1, b ^ 1);   // DMA next tile while computing
        v16bf a = load_frag_a(&Al[b][(16 * w + lo16) * GP], 0, hi);
#pragma unroll
        for (int j = 0; j < 8; ++j) {
            v16bf bb = load_frag_b2(&Bl[b][(16 * j + lo16) * GP], hi);
            acc[j] = wmma_bf16(a, bb, acc[j]);
        }
        wait_stage();
    }

#pragma unroll
    for (int j = 0; j < 8; ++j) {
#pragma unroll
        for (int r = 0; r < 8; ++r) {
            int m = m0 + 16 * w + r + 8 * hi;
            int n = n0 + 16 * j + lo16;
            C[(size_t)m * N + n] = acc[j][r];
        }
    }
}

// ---------------------------------------------------------------------------
// Llama3 RoPE frequency scaling (matches reference _rope_sin_cos)
// ---------------------------------------------------------------------------
__device__ __forceinline__ float llama_inv_freq(int d /*0..HD/2-1*/) {
    float expo     = (float)(2 * d) / (float)HD;
    float inv_freq = __powf(THETA, -expo);
    float wavelen  = 2.0f * 3.14159265358979323846f / inv_freq;
    float low_wl   = OLD_CTX / LOW_F;
    float high_wl  = OLD_CTX / HIGH_F;
    float inv_l    = (wavelen > low_wl) ? inv_freq / FACTOR : inv_freq;
    float smooth   = (OLD_CTX / wavelen - LOW_F) / (HIGH_F - LOW_F);
    float smoothed = (1.0f - smooth) * inv_l / FACTOR + smooth * inv_l;
    bool  med      = (wavelen >= high_wl) && (wavelen <= low_wl);
    return med ? smoothed : inv_l;
}

__global__ void rope_convert(const float* __restrict__ src, const int* __restrict__ pos,
                             bf16_t* __restrict__ dst, int heads, float scale) {
    int i = blockIdx.x * blockDim.x + threadIdx.x;
    int total = Ts * heads * (HD / 2);
    if (i >= total) return;
    int d = i & (HD / 2 - 1);
    int h = (i >> 5) % heads;
    int t = i / ((HD / 2) * heads);

    float f   = llama_inv_freq(d);
    float ang = (float)pos[t] * f;
    float s   = __sinf(ang);
    float c   = __cosf(ang);

    int stride = heads * HD;
    const float* p = src + (size_t)t * stride + h * HD;
    float x1 = p[d];
    float x2 = p[d + HD / 2];
    bf16_t* q = dst + (size_t)t * stride + h * HD;
    q[d]          = (bf16_t)((x1 * c - x2 * s) * scale);
    q[d + HD / 2] = (bf16_t)((x2 * c + x1 * s) * scale);
}

// ---------------------------------------------------------------------------
// Flash attention. Block = 8 waves = 128 queries of one head.
// Per 32-key step: K-tile (32x64) and V^T-tile (64x32) async-staged to LDS,
// shared by all 8 waves. Online softmax in f32; P re-swizzled via per-wave LDS.
//   qb: [T][NH*HD] bf16 (RoPE'd, pre-scaled by HD^-0.5)
//   kb: [T][KV*HD] bf16 (RoPE'd)
//   vt: [KV*HD][T] bf16 (transposed V)
//   ob: [T][NH*HD] bf16
// ---------------------------------------------------------------------------
__global__ void __launch_bounds__(256)
flash_attn(const bf16_t* __restrict__ qb, const bf16_t* __restrict__ kb,
           const bf16_t* __restrict__ vt, bf16_t* __restrict__ ob) {
    constexpr int KP = 72;   // K-tile row pitch (64-elem rows)
    constexpr int VP = 40;   // V-tile row pitch (32-elem rows)
    constexpr int PP = 40;   // P row pitch
    __shared__ bf16_t Klds[32 * KP];
    __shared__ bf16_t Vlds[64 * VP];
    __shared__ bf16_t Plds[8][16 * PP];

    const int tid  = threadIdx.x;
    const int w    = tid >> 5;
    const int lane = tid & 31;
    const int lo16 = lane & 15;
    const int hi   = lane >> 4;
    const int qBlk = blockIdx.x & (Ts / 128 - 1);   // 0..15
    const int h    = blockIdx.x >> 4;               // 0..31
    const int kvh  = h / Gg;
    const int qB   = qBlk << 7;
    const int q0   = qB + 16 * w;

    const bf16_t* qrow = qb + (size_t)(q0 + lo16) * NHD + h * HD;
    v16bf qf0 = load_frag_a(qrow, 0, hi);
    v16bf qf1 = load_frag_a(qrow, 32, hi);

    v8f o[4] = {v8f{}, v8f{}, v8f{}, v8f{}};
    float mst[8], lst[8];
#pragma unroll
    for (int r = 0; r < 8; ++r) { mst[r] = -3.0e38f; lst[r] = 0.0f; }

    const int kRow = tid >> 3, kOff = (tid & 7) * 8;   // K-tile staging coords
    const int vRow = tid >> 2, vOff = (tid & 3) * 8;   // V-tile staging coords

    for (int s0 = 0; s0 <= qB + 127; s0 += 32) {
        stage16(kb + (size_t)(s0 + kRow) * KVD + kvh * HD + kOff, &Klds[kRow * KP + kOff]);
        stage16(vt + (size_t)(kvh * HD + vRow) * Ts + s0 + vOff,  &Vlds[vRow * VP + vOff]);
        wait_stage();

        if (s0 <= q0 + 15) {
            // ---- logits: Q(16xHD) * K^T(HDx32) ----
            v8f sf[2] = {v8f{}, v8f{}};
#pragma unroll
            for (int j = 0; j < 2; ++j) {
                const bf16_t* krow = &Klds[(16 * j + lo16) * KP];
                sf[j] = wmma_bf16(qf0, load_frag_b2(krow, hi), sf[j]);
                sf[j] = wmma_bf16(qf1, load_frag_b2(krow + 32, hi), sf[j]);
            }

            // ---- causal mask + online softmax ----
#pragma unroll
            for (int r = 0; r < 8; ++r) {
                int t  = q0 + r + 8 * hi;
                float v0 = (s0 + lo16      <= t) ? sf[0][r] : -1.0e30f;
                float v1 = (s0 + 16 + lo16 <= t) ? sf[1][r] : -1.0e30f;
                float mx = fmaxf(v0, v1);
#pragma unroll
                for (int d = 1; d < 16; d <<= 1) mx = fmaxf(mx, __shfl_xor(mx, d, 16));
                float mnew  = fmaxf(mst[r], mx);
                float alpha = __expf(mst[r] - mnew);
                mst[r] = mnew;
                float p0 = __expf(v0 - mnew);
                float p1 = __expf(v1 - mnew);
                float ps = p0 + p1;
#pragma unroll
                for (int d = 1; d < 16; d <<= 1) ps += __shfl_xor(ps, d, 16);
                lst[r] = lst[r] * alpha + ps;
#pragma unroll
                for (int j = 0; j < 4; ++j) o[j][r] *= alpha;
                sf[0][r] = p0;
                sf[1][r] = p1;
            }

            // ---- re-swizzle P: C-layout -> A-layout via per-wave LDS ----
#pragma unroll
            for (int r = 0; r < 8; ++r) {
                int m = r + 8 * hi;
                Plds[w][m * PP + lo16]      = (bf16_t)sf[0][r];
                Plds[w][m * PP + 16 + lo16] = (bf16_t)sf[1][r];
            }
            lds_fence_wave();
            v16bf pf = load_frag_a(&Plds[w][lo16 * PP], 0, hi);

            // ---- O += P(16x32) * V(32x64) ----
#pragma unroll
            for (int j = 0; j < 4; ++j) {
                v16bf vf = load_frag_b2(&Vlds[(16 * j + lo16) * VP], hi);
                o[j] = wmma_bf16(pf, vf, o[j]);
            }
        }
        __syncthreads();   // all waves done with K/V tiles before next stage
    }

#pragma unroll
    for (int r = 0; r < 8; ++r) {
        float invl = 1.0f / lst[r];
        int t = q0 + r + 8 * hi;
#pragma unroll
        for (int j = 0; j < 4; ++j)
            ob[(size_t)t * NHD + h * HD + 16 * j + lo16] = (bf16_t)(o[j][r] * invl);
    }
}

// ---------------------------------------------------------------------------
// Host-side launcher
// ---------------------------------------------------------------------------
static inline size_t align256(size_t x) { return (x + 255) & ~(size_t)255; }

extern "C" void kernel_launch(void* const* d_in, const int* in_sizes, int n_in,
                              void* d_out, int out_size, void* d_ws, size_t ws_size,
                              hipStream_t stream) {
    const float* x   = (const float*)d_in[0];
    const int*   pos = (const int*)d_in[1];
    // d_in[2] = attn_mask (causal, reconstructed analytically)
    const float* wq  = (const float*)d_in[3];
    const float* wk  = (const float*)d_in[4];
    const float* wv  = (const float*)d_in[5];
    const float* wo  = (const float*)d_in[6];
    float* out = (float*)d_out;

    char* ws = (char*)d_ws;
    size_t off = 0;
    auto alloc = [&](size_t bytes) { char* p = ws + off; off += align256(bytes); return p; };
    bf16_t* xb  = (bf16_t*)alloc((size_t)Ts * Dd * 2);
    bf16_t* wqT = (bf16_t*)alloc((size_t)Dd * NHD * 2);
    bf16_t* wkT = (bf16_t*)alloc((size_t)Dd * KVD * 2);
    bf16_t* wvT = (bf16_t*)alloc((size_t)Dd * KVD * 2);
    bf16_t* woT = (bf16_t*)alloc((size_t)NHD * Dd * 2);
    float*  qf  = (float*)alloc((size_t)Ts * NHD * 4);
    float*  kf  = (float*)alloc((size_t)Ts * KVD * 4);
    float*  vf  = (float*)alloc((size_t)Ts * KVD * 4);
    bf16_t* qbf = (bf16_t*)alloc((size_t)Ts * NHD * 2);
    bf16_t* kbf = (bf16_t*)alloc((size_t)Ts * KVD * 2);
    bf16_t* vtb = (bf16_t*)alloc((size_t)KVD * Ts * 2);
    bf16_t* obf = (bf16_t*)alloc((size_t)Ts * NHD * 2);
    (void)ws_size; (void)n_in; (void)in_sizes; (void)out_size;

    const int CT = 256;
    cvt_f32_bf16<<<(Ts * Dd + CT - 1) / CT, CT, 0, stream>>>(x, xb, Ts * Dd);
    tr_f32_bf16<<<(Dd * NHD + CT - 1) / CT, CT, 0, stream>>>(wq, wqT, Dd, NHD);
    tr_f32_bf16<<<(Dd * KVD + CT - 1) / CT, CT, 0, stream>>>(wk, wkT, Dd, KVD);
    tr_f32_bf16<<<(Dd * KVD + CT - 1) / CT, CT, 0, stream>>>(wv, wvT, Dd, KVD);
    tr_f32_bf16<<<(NHD * Dd + CT - 1) / CT, CT, 0, stream>>>(wo, woT, NHD, Dd);

    // QKV projections (LDS-tiled WMMA GEMM, 128x128 blocks)
    gemm_bf16_lds<<<(Ts / 128) * (NHD / 128), CT, 0, stream>>>(xb, wqT, qf, Ts, NHD, Dd);
    gemm_bf16_lds<<<(Ts / 128) * (KVD / 128), CT, 0, stream>>>(xb, wkT, kf, Ts, KVD, Dd);
    gemm_bf16_lds<<<(Ts / 128) * (KVD / 128), CT, 0, stream>>>(xb, wvT, vf, Ts, KVD, Dd);

    // RoPE + bf16 convert (q pre-scaled by HD^-0.5); V transpose-convert
    rope_convert<<<(Ts * NH * (HD / 2) + CT - 1) / CT, CT, 0, stream>>>(qf, pos, qbf, NH, 0.125f);
    rope_convert<<<(Ts * KV * (HD / 2) + CT - 1) / CT, CT, 0, stream>>>(kf, pos, kbf, KV, 1.0f);
    tr_f32_bf16<<<(Ts * KVD + CT - 1) / CT, CT, 0, stream>>>(vf, vtb, Ts, KVD);

    // flash attention: 512 blocks x 256 threads (head x 128-query block)
    flash_attn<<<NH * (Ts / 128), CT, 0, stream>>>(qbf, kbf, vtb, obf);

    // output projection -> d_out f32
    gemm_bf16_lds<<<(Ts / 128) * (Dd / 128), CT, 0, stream>>>(obf, woT, out, Ts, Dd, NHD);
}
